// RNet_27711128994153
// MI455X (gfx1250) — matrix operand order repaired and checked by reading, not compiled, not run on hardware
//
#include <hip/hip_runtime.h>
#include <hip/hip_bf16.h>

typedef _Float16 half_t;
typedef __attribute__((ext_vector_type(16))) _Float16 v16h;
typedef __attribute__((ext_vector_type(8)))  _Float16 v8h;
typedef __attribute__((ext_vector_type(8)))  float    v8f;

#define C_   256
#define C1_  128
#define C2_  64
#define NR   256
#define BS   4
#define QD   1024
#define IND  2052

#define WMMA16(A, B, C) __builtin_amdgcn_wmma_f32_16x16x32_f16(false, (A), false, (B), (short)0, (C), false, false)

// ---------------- Qp = Q @ Wq + bq  -> [4,256] f32 ----------------
__global__ void qp_kernel(const float* __restrict__ Q, const float* __restrict__ Wq,
                          const float* __restrict__ bq, float* __restrict__ Qp) {
    const int b = blockIdx.x, c = threadIdx.x;
    const float* q = Q + b * QD;
    float acc = bq[c];
    for (int k = 0; k < QD; ++k) acc = fmaf(q[k], Wq[k * C_ + c], acc);
    Qp[b * C_ + c] = acc;
}

// ---- Xp = concat(X, X[:,:,:4]) @ Wv + bv + Qp -> f16 [4,256,256] ----
__global__ void xp_kernel(const float* __restrict__ X, const float* __restrict__ Wv,
                          const float* __restrict__ bv, const float* __restrict__ Qp,
                          half_t* __restrict__ Xph) {
    const int n = blockIdx.x, b = blockIdx.y, c = threadIdx.x;
    const float* xr = X + (size_t)(b * NR + n) * IND;
    float acc = bv[c] + Qp[b * C_ + c];
    for (int d = 0; d < IND; ++d) acc = fmaf(xr[d], Wv[d * C_ + c], acc);
    #pragma unroll
    for (int d = 0; d < 4; ++d)   acc = fmaf(xr[d], Wv[(IND + d) * C_ + c], acc);
    Xph[(size_t)(b * NR + n) * C_ + c] = (half_t)acc;
}

// --------- transpose weights to f16 [N][K] for WMMA B fragments ---------
__global__ void wtrans_kernel(const float* __restrict__ W01, const float* __restrict__ W1,
                              const float* __restrict__ W02, const float* __restrict__ W2,
                              half_t* __restrict__ WT1, half_t* __restrict__ WT2) {
    const int idx = blockIdx.x * 256 + threadIdx.x;
    if (idx < 2 * C1_ * C_) {
        const int br = idx / (C1_ * C_);
        const int r  = idx % (C1_ * C_);
        const int n = r / C_, c = r % C_;
        const float* src = br ? W1 : W01;      // [C][C1] row-major
        WT1[br * C1_ * C_ + n * C_ + c] = (half_t)src[c * C1_ + n];
    } else {
        const int idx2 = idx - 2 * C1_ * C_;
        if (idx2 < 2 * C2_ * C1_) {
            const int br = idx2 / (C2_ * C1_);
            const int r  = idx2 % (C2_ * C1_);
            const int n = r / C1_, c = r % C1_;
            const float* src = br ? W2 : W02;  // [C1][C2] row-major
            WT2[br * C2_ * C1_ + n * C1_ + c] = (half_t)src[c * C2_ + n];
        }
    }
}

// ---------------- pairwise 3-layer MLP, both branches ----------------
// grid (16 j-blocks, 16 i-blocks, 4 batches), 256 threads = 8 waves (wave32)
__global__ void __launch_bounds__(256, 2)
pair_mlp_kernel(const half_t* __restrict__ Xph,
                const half_t* __restrict__ WT1,  // [2][128][256] f16
                const half_t* __restrict__ WT2,  // [2][64][128]  f16
                const float* __restrict__ b01, const float* __restrict__ b1,
                const float* __restrict__ b02, const float* __restrict__ b2,
                const float* __restrict__ W03, const float* __restrict__ W3,
                const float* __restrict__ b03, const float* __restrict__ b3,
                float* __restrict__ hbuf) {     // [2][4][256][256] f32
    extern __shared__ char smem[];
    half_t* XpI = (half_t*)smem;            // 16*256 f16 =  8 KB
    half_t* XpJ = XpI + 16 * C_;            // 16*256 f16 =  8 KB
    half_t* H1  = XpJ + 16 * C_;            // 256*128 f16 = 64 KB
    half_t* H2  = H1 + NR * C1_;            // 256*64  f16 = 32 KB

    const int tid  = threadIdx.x;
    const int b    = blockIdx.z;
    const int i0   = blockIdx.y * 16;
    const int j0   = blockIdx.x * 16;
    const int wave = tid >> 5;
    const int lane = tid & 31;
    const int li   = lane & 15;
    const int hf   = lane >> 4;

    // stage 16 i-rows and 16 j-rows of Xp (f16) into LDS
    {
        const int row = tid >> 4;
        const int c0  = (tid & 15) * 16;
        *(v16h*)(XpI + row * C_ + c0) =
            *(const v16h*)(Xph + (size_t)(b * NR + i0 + row) * C_ + c0);
        *(v16h*)(XpJ + row * C_ + c0) =
            *(const v16h*)(Xph + (size_t)(b * NR + j0 + row) * C_ + c0);
    }
    __syncthreads();

    union Frag { v16h v; v8h h[2]; };

    for (int br = 0; br < 2; ++br) {
        const half_t* w1t   = WT1 + br * (C1_ * C_);
        const half_t* w2t   = WT2 + br * (C2_ * C1_);
        const float*  bias1 = br ? b1 : b01;
        const float*  bias2 = br ? b2 : b02;
        const float*  w3    = br ? W3 : W03;
        const float   bias3 = br ? b3[0] : b03[0];

        // ---- GEMM1: A = P-tile [256 pairs x 256], B = W1 [256 x 128] ----
        // wave owns TWO N-tiles (N = 32*wq .. 32*wq+31) and half the M-tiles.
        // A fragment double-buffered, CONSUME-then-BUILD order so the other
        // buffer's 8 pk_muls fill the WMMA->VALU WAR hazard slots.
        {
            const int wq     = wave & 3;        // N-pair 0..3
            const int tbase  = (wave >> 2) * 8; // M-tiles [tbase, tbase+8)
            v16h B1a[8], B1b[8];
            const half_t* wpa = w1t + (wq * 32 + li) * C_ + hf * 8;
            const half_t* wpb = wpa + 16 * C_;
            #pragma unroll
            for (int s = 0; s < 8; ++s) { B1a[s] = __builtin_shufflevector(*(const v8h*)(wpa + 32 * s), *(const v8h*)(wpa + 32 * s + 16), 0,1,2,3,4,5,6,7,8,9,10,11,12,13,14,15);
                                          B1b[s] = __builtin_shufflevector(*(const v8h*)(wpb + 32 * s), *(const v8h*)(wpb + 32 * s + 16), 0,1,2,3,4,5,6,7,8,9,10,11,12,13,14,15); }
            const float biasA = bias1[wq * 32 + li];
            const float biasB = bias1[wq * 32 + 16 + li];
            #pragma unroll 1
            for (int tt = 0; tt < 8; ++tt) {
                const int t = tbase + tt;       // pairs (i0+t, j0+0..15)
                v8f accA, accB;
                #pragma unroll
                for (int r = 0; r < 8; ++r) { accA[r] = biasA; accB[r] = biasB; }
                const half_t* xi = XpI + t  * C_ + hf * 8;
                const half_t* xj = XpJ + li * C_ + hf * 8;
                Frag a0, a1;
                a0.h[0] = *(const v8h*)(xi)      * *(const v8h*)(xj);
                a0.h[1] = *(const v8h*)(xi + 16) * *(const v8h*)(xj + 16);
                #pragma unroll
                for (int s = 0; s < 8; ++s) {
                    const Frag& cur = (s & 1) ? a1 : a0;
                    Frag&       nxt = (s & 1) ? a0 : a1;
                    accA = WMMA16(cur.v, B1a[s], accA);
                    accB = WMMA16(cur.v, B1b[s], accB);
                    if (s < 7) {   // build next fragment after consuming current
                        nxt.h[0] = *(const v8h*)(xi + 32 * (s + 1))
                                 * *(const v8h*)(xj + 32 * (s + 1));
                        nxt.h[1] = *(const v8h*)(xi + 32 * (s + 1) + 16)
                                 * *(const v8h*)(xj + 32 * (s + 1) + 16);
                    }
                }
                #pragma unroll
                for (int r = 0; r < 8; ++r) {
                    float vA = accA[r] > 0.f ? accA[r] : 0.f;   // ReLU
                    float vB = accB[r] > 0.f ? accB[r] : 0.f;
                    H1[(t * 16 + hf * 8 + r) * C1_ + wq * 32 + li]      = (half_t)vA;
                    H1[(t * 16 + hf * 8 + r) * C1_ + wq * 32 + 16 + li] = (half_t)vB;
                }
            }
        }
        __syncthreads();

        // ---- GEMM2: A = H1 [256 x 128], B = W2 [128 x 64] ----
        {
            const int np    = wave & 1;         // N-pair 0..1 (N = 32*np..32*np+31)
            const int tbase = (wave >> 1) * 4;  // M-tiles [tbase, tbase+4)
            v16h B2a[4], B2b[4];
            const half_t* wpa = w2t + (np * 32 + li) * C1_ + hf * 8;
            const half_t* wpb = wpa + 16 * C1_;
            #pragma unroll
            for (int s = 0; s < 4; ++s) { B2a[s] = __builtin_shufflevector(*(const v8h*)(wpa + 32 * s), *(const v8h*)(wpa + 32 * s + 16), 0,1,2,3,4,5,6,7,8,9,10,11,12,13,14,15);
                                          B2b[s] = __builtin_shufflevector(*(const v8h*)(wpb + 32 * s), *(const v8h*)(wpb + 32 * s + 16), 0,1,2,3,4,5,6,7,8,9,10,11,12,13,14,15); }
            const float biasA = bias2[np * 32 + li];
            const float biasB = bias2[np * 32 + 16 + li];
            #pragma unroll 1
            for (int tt = 0; tt < 4; ++tt) {
                const int t = tbase + tt;
                v8f accA, accB;
                #pragma unroll
                for (int r = 0; r < 8; ++r) { accA[r] = biasA; accB[r] = biasB; }
                const half_t* ar = H1 + (t * 16 + li) * C1_ + hf * 8;
                Frag a0, a1;
                a0.h[0] = *(const v8h*)(ar);
                a0.h[1] = *(const v8h*)(ar + 16);
                #pragma unroll
                for (int s = 0; s < 4; ++s) {
                    const Frag& cur = (s & 1) ? a1 : a0;
                    Frag&       nxt = (s & 1) ? a0 : a1;
                    accA = WMMA16(cur.v, B2a[s], accA);
                    accB = WMMA16(cur.v, B2b[s], accB);
                    if (s < 3) {
                        nxt.h[0] = *(const v8h*)(ar + 32 * (s + 1));
                        nxt.h[1] = *(const v8h*)(ar + 32 * (s + 1) + 16);
                    }
                }
                #pragma unroll
                for (int r = 0; r < 8; ++r) {
                    float vA = accA[r] > 0.f ? accA[r] : 0.f;
                    float vB = accB[r] > 0.f ? accB[r] : 0.f;
                    H2[(t * 16 + hf * 8 + r) * C2_ + np * 32 + li]      = (half_t)vA;
                    H2[(t * 16 + hf * 8 + r) * C2_ + np * 32 + 16 + li] = (half_t)vB;
                }
            }
        }
        __syncthreads();

        // ---- layer3: [256 x 64] x [64 x 1], ReLU, write h map ----
        {
            float acc = bias3;
            const half_t* hr = H2 + tid * C2_;
            #pragma unroll
            for (int k = 0; k < C2_; ++k) acc = fmaf((float)hr[k], w3[k], acc);
            const float v = acc > 0.f ? acc : 0.f;
            const int i = i0 + (tid >> 4);
            const int j = j0 + (tid & 15);
            hbuf[((size_t)(br * BS + b) * NR + i) * NR + j] = v;
        }
        __syncthreads();
    }
}

// -------- per (branch,batch): global max and sum-exp of s = h + h^T --------
__global__ void stats_kernel(const float* __restrict__ hbuf, float* __restrict__ stats) {
    __shared__ float red[256];
    const int br = blockIdx.x >> 2;
    const int b  = blockIdx.x & 3;
    const int tid = threadIdx.x;
    const float* hb = hbuf + (size_t)(br * BS + b) * NR * NR;
    float m = -1e30f;
    for (int k = 0; k < NR; ++k)
        m = fmaxf(m, hb[k * NR + tid] + hb[tid * NR + k]);
    red[tid] = m; __syncthreads();
    for (int off = 128; off > 0; off >>= 1) {
        if (tid < off) red[tid] = fmaxf(red[tid], red[tid + off]);
        __syncthreads();
    }
    const float gmax = red[0];
    __syncthreads();
    float z = 0.f;
    for (int k = 0; k < NR; ++k)
        z += __expf(hb[k * NR + tid] + hb[tid * NR + k] - gmax);
    red[tid] = z; __syncthreads();
    for (int off = 128; off > 0; off >>= 1) {
        if (tid < off) red[tid] += red[tid + off];
        __syncthreads();
    }
    if (tid == 0) { stats[blockIdx.x * 2] = gmax; stats[blockIdx.x * 2 + 1] = red[0]; }
}

// -------- w[b,j] = 0.5 * sum_i (m0[i,j] + m1[i,j]) --------
__global__ void colsum_kernel(const float* __restrict__ hbuf, const float* __restrict__ stats,
                              float* __restrict__ wout) {
    const int b = blockIdx.x, j = threadIdx.x;
    const float m0 = stats[(0 * BS + b) * 2], z0 = stats[(0 * BS + b) * 2 + 1];
    const float m1 = stats[(1 * BS + b) * 2], z1 = stats[(1 * BS + b) * 2 + 1];
    const float* h0 = hbuf + (size_t)(0 * BS + b) * NR * NR;
    const float* h1 = hbuf + (size_t)(1 * BS + b) * NR * NR;
    float a0 = 0.f, a1 = 0.f;
    for (int i = 0; i < NR; ++i) {
        a0 += __expf(h0[i * NR + j] + h0[j * NR + i] - m0);
        a1 += __expf(h1[i * NR + j] + h1[j * NR + i] - m1);
    }
    wout[b * NR + j] = 0.5f * (a0 / z0 + a1 / z1);
}

// -------- out[b,d] = sum_j w[b,j] * X[b,j,d] --------
__global__ void out_kernel(const float* __restrict__ X, const float* __restrict__ wsum,
                           float* __restrict__ out) {
    __shared__ float wsh[NR];
    const int b = blockIdx.y;
    wsh[threadIdx.x] = wsum[b * NR + threadIdx.x];
    __syncthreads();
    const int d = blockIdx.x * 256 + threadIdx.x;
    if (d < IND) {
        float acc = 0.f;
        for (int j = 0; j < NR; ++j)
            acc = fmaf(wsh[j], X[(size_t)(b * NR + j) * IND + d], acc);
        out[b * IND + d] = acc;
    }
}

extern "C" void kernel_launch(void* const* d_in, const int* in_sizes, int n_in,
                              void* d_out, int out_size, void* d_ws, size_t ws_size,
                              hipStream_t stream) {
    const float* Q   = (const float*)d_in[0];
    const float* X   = (const float*)d_in[1];
    const float* Wv  = (const float*)d_in[2];
    const float* bv  = (const float*)d_in[3];
    const float* Wq  = (const float*)d_in[4];
    const float* bq  = (const float*)d_in[5];
    const float* W01 = (const float*)d_in[6];
    const float* b01 = (const float*)d_in[7];
    const float* W02 = (const float*)d_in[8];
    const float* b02 = (const float*)d_in[9];
    const float* W03 = (const float*)d_in[10];
    const float* b03 = (const float*)d_in[11];
    const float* W1  = (const float*)d_in[12];
    const float* b1  = (const float*)d_in[13];
    const float* W2  = (const float*)d_in[14];
    const float* b2  = (const float*)d_in[15];
    const float* W3  = (const float*)d_in[16];
    const float* b3  = (const float*)d_in[17];

    char* ws = (char*)d_ws;
    float*  Qp    = (float*)(ws + 0);                      //   4 KB
    half_t* Xph   = (half_t*)(ws + 4096);                  // 512 KB
    half_t* WT1   = (half_t*)(ws + 4096 + 524288);         // 128 KB
    half_t* WT2   = (half_t*)(ws + 4096 + 524288 + 131072);//  32 KB
    float*  hbuf  = (float*)(ws + 692224);                 //   2 MB
    float*  stats = (float*)(ws + 2789376);                //  64 B
    float*  wsum  = (float*)(ws + 2789632);                //   4 KB

    qp_kernel<<<dim3(BS), dim3(256), 0, stream>>>(Q, Wq, bq, Qp);
    xp_kernel<<<dim3(NR, BS), dim3(256), 0, stream>>>(X, Wv, bv, Qp, Xph);
    wtrans_kernel<<<dim3(320), dim3(256), 0, stream>>>(W01, W1, W02, W2, WT1, WT2);

    const size_t shmem = (size_t)(16 * C_ + 16 * C_ + NR * C1_ + NR * C2_) * sizeof(half_t); // 112 KB
    pair_mlp_kernel<<<dim3(16, 16, BS), dim3(256), shmem, stream>>>(
        Xph, WT1, WT2, b01, b1, b02, b2, W03, W3, b03, b3, hbuf);

    stats_kernel<<<dim3(8), dim3(256), 0, stream>>>(hbuf, stats);
    colsum_kernel<<<dim3(BS), dim3(256), 0, stream>>>(hbuf, stats, wsum);
    out_kernel<<<dim3(9, BS), dim3(256), 0, stream>>>(X, wsum, (float*)d_out);
}
